// GraphConvolution_65274912964653
// MI455X (gfx1250) — matrix operand order repaired and checked by reading, not compiled, not run on hardware
//
#include <hip/hip_runtime.h>

#define IN_F 256
#define OUT_F 128

typedef __attribute__((ext_vector_type(16))) __bf16 v16bf;
typedef __attribute__((ext_vector_type(8)))  __bf16 v8bf;
typedef __attribute__((ext_vector_type(8)))  float  v8f;
typedef __attribute__((ext_vector_type(4)))  float  v4f;

// ---------------------------------------------------------------------------
// Swizzle fp32 weight [256][128] -> bf16 in WMMA B-fragment layout.
// B layout (16-bit, 32x16 KxN per k-step): lane l holds column n = l&15,
// K = kstep*32 + (l>>4)*16 + j for j=0..15, packed 2-per-VGPR in order.
// Bswz index: (((ks*8 + ntile)*32 + lane)*16 + j)
// ---------------------------------------------------------------------------
__global__ __launch_bounds__(256) void k_swizzle_w(const float* __restrict__ w,
                                                   __bf16* __restrict__ bswz) {
  const int idx  = blockIdx.x * blockDim.x + threadIdx.x;  // 0..2047
  const int lane = idx & 31;
  const int nt   = (idx >> 5) & 7;  // n-tile
  const int ks   = idx >> 8;        // k-step
  const int n    = nt * 16 + (lane & 15);
  const int kb   = ks * 32 + (lane >> 4) * 16;
  __bf16* dst = bswz + idx * 16;
#pragma unroll
  for (int j = 0; j < 16; ++j)
    dst[j] = (__bf16)w[(kb + j) * OUT_F + n];
}

// ---------------------------------------------------------------------------
// support = x @ W  via v_wmma_f32_16x16x32_bf16.
// Block = 256 threads = 8 waves. Block covers a 16-row M tile; wave w owns
// the 16-col N tile w. The 16x256 x-tile is staged in LDS as bf16 row-major.
// A fragment (16-bit, 16x32): lane l, row m = l&15,
//   elements 0..7  = K = ks*32 + (l>>4)*16 + 0..7
//   elements 8..15 = K = ks*32 + (l>>4)*16 + 8 + 0..7
// C/D (32-bit 16x16): VGPR v, lane l -> m = v + 8*(l>>4), n = l&15.
// ---------------------------------------------------------------------------
__global__ __launch_bounds__(256) void k_gemm(const float* __restrict__ x,
                                              const __bf16* __restrict__ bswz,
                                              float* __restrict__ support) {
  __shared__ __bf16 lds_a[16 * IN_F];  // 8 KB
  const int tid  = threadIdx.x;
  const int wave = tid >> 5;  // n-tile 0..7
  const int lane = tid & 31;
  const int m0   = blockIdx.x * 16;

  // Cooperatively stage 16x256 fp32 tile of x as bf16 into LDS (row-major).
  {
    const int row = tid >> 4;          // 0..15
    const int col = (tid & 15) * 16;   // 0..240
    const float* src = x + (size_t)(m0 + row) * IN_F + col;
    v4f f0 = *(const v4f*)(src + 0);
    v4f f1 = *(const v4f*)(src + 4);
    v4f f2 = *(const v4f*)(src + 8);
    v4f f3 = *(const v4f*)(src + 12);
    v16bf a;
#pragma unroll
    for (int i = 0; i < 4; ++i) {
      a[i]      = (__bf16)f0[i];
      a[i + 4]  = (__bf16)f1[i];
      a[i + 8]  = (__bf16)f2[i];
      a[i + 12] = (__bf16)f3[i];
    }
    *(v16bf*)(lds_a + row * IN_F + col) = a;  // 32B aligned
  }
  __syncthreads();

  const int hi   = lane >> 4;
  const int mrow = lane & 15;
  v8f acc = {};
#pragma unroll
  for (int ks = 0; ks < 8; ++ks) {
    // A fragment from LDS: two 16B reads of 8 consecutive bf16 K-values.
    const __bf16* ap = lds_a + mrow * IN_F + ks * 32 + hi * 16;
    v8bf alo = *(const v8bf*)(ap);
    v8bf ahi = *(const v8bf*)(ap + 8);
    v16bf a;
#pragma unroll
    for (int i = 0; i < 8; ++i) { a[i] = alo[i]; a[i + 8] = ahi[i]; }
    // B fragment: one 32B load from the pre-swizzled weight (L1-resident).
    v16bf b = *(const v16bf*)(bswz + (size_t)(((ks << 3) | wave) * 32 + lane) * 16);
    acc = __builtin_amdgcn_wmma_f32_16x16x32_bf16(false, a, false, b,
                                                  (short)0, acc, false, false);
  }
  float* op = support + (size_t)(m0 + hi * 8) * OUT_F + wave * 16 + mrow;
#pragma unroll
  for (int v = 0; v < 8; ++v) op[(size_t)v * OUT_F] = acc[v];
}

// ---------------------------------------------------------------------------
// out[n][f] = bias[f]
// ---------------------------------------------------------------------------
__global__ __launch_bounds__(256) void k_bias_init(float* __restrict__ out,
                                                   const float* __restrict__ bias,
                                                   int total) {
  const int i = blockIdx.x * 256 + threadIdx.x;
  if (i < total) out[i] = bias[i & (OUT_F - 1)];
}

// ---------------------------------------------------------------------------
// Edge-parallel scatter: wave per edge, 4 features per lane.
// support fits in the 192MB L2, so gathers and float atomics stay on-chip.
// ---------------------------------------------------------------------------
__global__ __launch_bounds__(256) void k_scatter(const int* __restrict__ erow,
                                                 const int* __restrict__ ecol,
                                                 const float* __restrict__ eval,
                                                 const float* __restrict__ support,
                                                 float* __restrict__ out,
                                                 int n_edges) {
  const int t    = blockIdx.x * 256 + threadIdx.x;
  const int e    = t >> 5;
  const int lane = t & 31;
  if (e >= n_edges) return;
  const int   r = erow[e];
  const int   c = ecol[e];
  const float v = eval[e];
  const v4f s = *(const v4f*)(support + (size_t)c * OUT_F + lane * 4);
  float* o = out + (size_t)r * OUT_F + lane * 4;
  unsafeAtomicAdd(o + 0, s[0] * v);
  unsafeAtomicAdd(o + 1, s[1] * v);
  unsafeAtomicAdd(o + 2, s[2] * v);
  unsafeAtomicAdd(o + 3, s[3] * v);
}

// ---------------------------------------------------------------------------
// relu in place
// ---------------------------------------------------------------------------
__global__ __launch_bounds__(256) void k_relu(float* __restrict__ out, int total) {
  const int i = blockIdx.x * 256 + threadIdx.x;
  if (i < total) out[i] = fmaxf(out[i], 0.0f);
}

extern "C" void kernel_launch(void* const* d_in, const int* in_sizes, int n_in,
                              void* d_out, int out_size, void* d_ws, size_t ws_size,
                              hipStream_t stream) {
  const float* x    = (const float*)d_in[0];
  const int*   erow = (const int*)d_in[1];
  const int*   ecol = (const int*)d_in[2];
  const float* eval = (const float*)d_in[3];
  const float* w    = (const float*)d_in[4];
  const float* bias = (const float*)d_in[5];
  float* out = (float*)d_out;

  const int n_nodes = in_sizes[0] / IN_F;  // 100000
  const int n_edges = in_sizes[1];         // 3200000

  // ws layout: [0, 64KB) bf16 swizzled weight; [64KB, ...) fp32 support.
  __bf16* bswz    = (__bf16*)d_ws;
  float*  support = (float*)((char*)d_ws + 65536);

  // 1) weight -> bf16 WMMA-B layout (8*8*32 = 2048 fragments)
  k_swizzle_w<<<8, 256, 0, stream>>>(w, bswz);

  // 2) support = x @ W  (n_nodes is a multiple of 16)
  k_gemm<<<n_nodes / 16, 256, 0, stream>>>(x, bswz, support);

  // 3) out = bias (broadcast)
  const int total = n_nodes * OUT_F;  // 12.8M
  k_bias_init<<<(total + 255) / 256, 256, 0, stream>>>(out, bias, total);

  // 4) out[row[e]] += val[e] * support[col[e]]
  const long long threads = (long long)n_edges * 32;
  k_scatter<<<(int)((threads + 255) / 256), 256, 0, stream>>>(erow, ecol, eval,
                                                              support, out, n_edges);

  // 5) relu
  k_relu<<<(total + 255) / 256, 256, 0, stream>>>(out, total);
}
